// Model_2886218023415
// MI455X (gfx1250) — compile-verified, matrix operand order checked by reading, not run on hardware
//
#include <hip/hip_runtime.h>

// ---------------------------------------------------------------------------
// MI455X (gfx1250, wave32) implementation.
// Heavy GEMMs -> v_wmma_f32_16x16x32_f16 with batch as the M dimension.
// All GEMM weights pre-converted to f16 (padded) so the sequential LSTM
// critical path issues only f16 b128 loads + WMMA (no f32->f16 cvt chains).
// Gate nonlinearities use the gfx1250 hardware v_tanh_f32 transcendental
// (sigmoid(x) = 0.5*tanh(x/2)+0.5) to keep the serial recurrence short.
// Intermediates in f16 (3 x 84MB buffers ~ L2 resident), f32 accumulation.
// ---------------------------------------------------------------------------

typedef __attribute__((ext_vector_type(16))) _Float16 v16h;
typedef __attribute__((ext_vector_type(8)))  float    v8f;

#define SEQN 10
#define NCHN 7
#define GHN  32

__device__ constexpr int ESRC[20] = {0,7,1,7,2,7,3,7,4,8,5,8,6,8,7,8,7,9,8,9};
__device__ constexpr int EDST[20] = {7,0,7,1,7,2,7,3,8,4,8,5,8,6,8,7,9,7,9,8};

// gfx1250 hardware tanh (TRANS32). Trailing v_nop satisfies the ISA rule:
// one independent op/V_NOP after a TRANS op before its output is consumed
// (the compiler cannot classify inline asm as TRANS for hazard insertion).
__device__ __forceinline__ float tanh_hw(float x){
  float r;
  asm volatile("v_tanh_f32 %0, %1\n\tv_nop" : "=v"(r) : "v"(x));
  return r;
}
__device__ __forceinline__ float sigmoid_hw(float x){
  return 0.5f * tanh_hw(0.5f * x) + 0.5f;
}
__device__ __forceinline__ float lrelu_(float x){ return x>0.f? x : 0.2f*x; }

// K-offset for f16 16x32 A/B fragment element group i (VGPR i), per ISA layout:
// lanes 0-15: VGPR0-3 -> K=0..7, VGPR4-7 -> K=16..23 ; lanes 16-31: +8
__device__ __forceinline__ int ak0_(int lane, int i){
  int k = (i < 4) ? (2*i) : (16 + 2*(i-4));
  return (lane & 16) ? (k + 8) : k;
}

template <typename F>
__device__ __forceinline__ v16h make_frag(F f){
  int lane = threadIdx.x & 31;
  int r16  = lane & 15;
  v16h r;
#pragma unroll
  for (int i = 0; i < 8; ++i){
    int k = ak0_(lane, i);
    r[2*i]   = f(r16, k);
    r[2*i+1] = f(r16, k+1);
  }
  return r;
}

__device__ __forceinline__ v8f wmma_f16f32(v16h a, v16h b, v8f c){
  return __builtin_amdgcn_wmma_f32_16x16x32_f16(false, a, false, b, (short)0, c, false, false);
}

// ---------------------------------------------------------------------------
// Weight conversion: f32 [rows][src_cols] -> f16 [rows][dst_cols], zero-padded
// (both in columns and in rows beyond `rows`, up to dst_rows).
// ---------------------------------------------------------------------------
__global__ void convert_w_kernel(const float* __restrict__ src, _Float16* __restrict__ dst,
                                 int rows, int dst_rows, int src_cols, int dst_cols){
  int idx = blockIdx.x * blockDim.x + threadIdx.x;
  int total = dst_rows * dst_cols;
  if (idx >= total) return;
  int r = idx / dst_cols, c = idx % dst_cols;
  float v = (r < rows && c < src_cols) ? src[r * src_cols + c] : 0.f;
  dst[idx] = (_Float16)v;
}

// ---------------------------------------------------------------------------
// Prep: fold GAT attention vectors into the weight matrices (tiny).
// prep[0..39]   P_as1[f][h], prep[40..79] P_ad1[f][h]
// prep[80..207] u2[c],       prep[208..335] v2[c]
// ---------------------------------------------------------------------------
__global__ void prep_kernel(const float* __restrict__ ga1_W, const float* __restrict__ ga1_as,
                            const float* __restrict__ ga1_ad,
                            const float* __restrict__ ga2_W, const float* __restrict__ ga2_as,
                            const float* __restrict__ ga2_ad,
                            float* __restrict__ prep){
  int t = threadIdx.x;
  if (t < 40){
    int f = t / 4, h = t % 4;
    float s = 0.f, d = 0.f;
    for (int c = 0; c < 32; ++c){
      float w = ga1_W[f*128 + h*32 + c];
      s += w * ga1_as[h*32 + c];
      d += w * ga1_ad[h*32 + c];
    }
    prep[t] = s; prep[40 + t] = d;
  } else if (t >= 64 && t < 192){
    int c = t - 64;
    float u = 0.f, v = 0.f;
    for (int s10 = 0; s10 < 10; ++s10){
      float w = ga2_W[c*10 + s10];
      u += w * ga2_as[s10];
      v += w * ga2_ad[s10];
    }
    prep[80 + c] = u; prep[208 + c] = v;
  }
}

// ---------------------------------------------------------------------------
// Graph front-end: GAT1 -> GAT2 (Mp) and GraphConv1 -> GraphConv2 (bfeat).
// One thread per sample; per-thread scratch in LDS ([idx][tid] layout).
// Output x1: f16 [B][10][32]  (cols 0-6 M^T, 7-13 Mp^T, 14-20 bfeat^T, 21-31 zero)
// ---------------------------------------------------------------------------
__global__ __launch_bounds__(32)
void graph_kernel(const float* __restrict__ M,
                  const float* __restrict__ W1,   const float* __restrict__ b1,
                  const float* __restrict__ W2,   const float* __restrict__ b2,
                  const float* __restrict__ Wrel1,const float* __restrict__ Wroot1, const float* __restrict__ gb1,
                  const float* __restrict__ Wrel2,const float* __restrict__ Wroot2, const float* __restrict__ gb2,
                  const float* __restrict__ prep,
                  _Float16* __restrict__ x1, int B){
  __shared__ float sA[100][32];   // GAT alpha, later GraphConv agg
  __shared__ float sX[100][32];   // GAT2 xh2, later bfeat (first 70)
  __shared__ float sMP[70][32];   // Mp
  int tid = threadIdx.x;
  long b = (long)blockIdx.x * 32 + tid;
  if (b >= B) return;
  const float* Mb  = M + b * 70;
  const float* Pas = prep, *Pad = prep + 40, *u2 = prep + 80, *v2 = prep + 208;

  float as2[10], ad2[10];
#pragma unroll
  for (int n = 0; n < 10; ++n){ as2[n] = 0.f; ad2[n] = 0.f; }
  for (int i = 0; i < 100; ++i) sX[i][tid] = 0.f;

  // ---- GAT layer 1 (4 heads), stream over output channels ----
  for (int h = 0; h < 4; ++h){
    float av[10], dv[10];
#pragma unroll
    for (int n = 0; n < 10; ++n){
      float s = 0.f, d = 0.f;
      if (n < 7){
#pragma unroll
        for (int f = 0; f < 10; ++f){ float v = Mb[n*10+f]; s += v*Pas[f*4+h]; d += v*Pad[f*4+h]; }
      }
      av[n] = s; dv[n] = d;
    }
    float e[30], mx[10], den[10];
#pragma unroll
    for (int j = 0; j < 20; ++j) e[j] = lrelu_(av[ESRC[j]] + dv[EDST[j]]);
#pragma unroll
    for (int n = 0; n < 10; ++n) e[20+n] = lrelu_(av[n] + dv[n]);
#pragma unroll
    for (int n = 0; n < 10; ++n){ mx[n] = -1e30f; den[n] = 0.f; }
#pragma unroll
    for (int j = 0; j < 20; ++j) mx[EDST[j]] = fmaxf(mx[EDST[j]], e[j]);
#pragma unroll
    for (int n = 0; n < 10; ++n) mx[n] = fmaxf(mx[n], e[20+n]);
#pragma unroll
    for (int j = 0; j < 20; ++j){ e[j] = __expf(e[j] - mx[EDST[j]]); den[EDST[j]] += e[j]; }
#pragma unroll
    for (int n = 0; n < 10; ++n){ e[20+n] = __expf(e[20+n] - mx[n]); den[n] += e[20+n]; }
    for (int i = 0; i < 100; ++i) sA[i][tid] = 0.f;
#pragma unroll
    for (int j = 0; j < 20; ++j) sA[EDST[j]*10 + ESRC[j]][tid] = e[j] / den[EDST[j]];
#pragma unroll
    for (int n = 0; n < 10; ++n) sA[n*11][tid] = e[20+n] / den[n];

    for (int c = 0; c < GHN; ++c){
      int cg = h*GHN + c;
      float xcol[10];
#pragma unroll
      for (int m = 0; m < 10; ++m){
        float s = 0.f;
        if (m < 7){
#pragma unroll
          for (int f = 0; f < 10; ++f) s += Mb[m*10+f] * W1[f*128 + cg];
        }
        xcol[m] = s;
      }
      float bb = b1[cg], uu = u2[cg], vv = v2[cg];
#pragma unroll
      for (int n = 0; n < 10; ++n){
        float a = 0.f;
#pragma unroll
        for (int m = 0; m < 10; ++m) a += sA[n*10+m][tid] * xcol[m];
        float hv = fmaxf(a + bb, 0.f);         // relu(gat1)
        as2[n] += hv * uu; ad2[n] += hv * vv;
#pragma unroll
        for (int s10 = 0; s10 < 10; ++s10) sX[n*10+s10][tid] += hv * W2[cg*10 + s10];
      }
    }
  }

  // ---- GAT layer 2 attention + aggregation (Mp, nodes 0..6) ----
  {
    float e2[30], mx2[10], den2[10];
#pragma unroll
    for (int j = 0; j < 20; ++j) e2[j] = lrelu_(as2[ESRC[j]] + ad2[EDST[j]]);
#pragma unroll
    for (int n = 0; n < 10; ++n) e2[20+n] = lrelu_(as2[n] + ad2[n]);
#pragma unroll
    for (int n = 0; n < 10; ++n){ mx2[n] = -1e30f; den2[n] = 0.f; }
#pragma unroll
    for (int j = 0; j < 20; ++j) mx2[EDST[j]] = fmaxf(mx2[EDST[j]], e2[j]);
#pragma unroll
    for (int n = 0; n < 10; ++n) mx2[n] = fmaxf(mx2[n], e2[20+n]);
#pragma unroll
    for (int j = 0; j < 20; ++j){ e2[j] = __expf(e2[j] - mx2[EDST[j]]); den2[EDST[j]] += e2[j]; }
#pragma unroll
    for (int n = 0; n < 10; ++n){ e2[20+n] = __expf(e2[20+n] - mx2[n]); den2[n] += e2[20+n]; }
    for (int i = 0; i < 70; ++i) sMP[i][tid] = 0.f;
#pragma unroll
    for (int j = 0; j < 20; ++j){
      if (EDST[j] < 7){
        float al = e2[j] / den2[EDST[j]];
#pragma unroll
        for (int s10 = 0; s10 < 10; ++s10)
          sMP[EDST[j]*10 + s10][tid] += al * sX[ESRC[j]*10 + s10][tid];
      }
    }
#pragma unroll
    for (int n = 0; n < 7; ++n){
      float al = e2[20+n] / den2[n];
#pragma unroll
      for (int s10 = 0; s10 < 10; ++s10) sMP[n*10 + s10][tid] += al * sX[n*10 + s10][tid];
    }
  }

  // ---- GraphConv 1 & 2 (no self loops). sA := agg(nf), sX[0..69] := bfeat ----
  for (int i = 0; i < 100; ++i) sA[i][tid] = 0.f;
#pragma unroll
  for (int j = 0; j < 20; ++j){
    if (ESRC[j] < 7){
#pragma unroll
      for (int f = 0; f < 10; ++f) sA[EDST[j]*10 + f][tid] += Mb[ESRC[j]*10 + f];
    }
  }
  for (int i = 0; i < 70; ++i) sX[i][tid] = gb2[i % 10];
  for (int c = 0; c < GHN; ++c){
    float gcol[10], gagg[10];
#pragma unroll
    for (int n = 0; n < 10; ++n) gagg[n] = 0.f;
#pragma unroll
    for (int n = 0; n < 10; ++n){
      float a = 0.f;
#pragma unroll
      for (int f = 0; f < 10; ++f) a += sA[n*10+f][tid] * Wrel1[f*32 + c];
      float rt = 0.f;
      if (n < 7){
#pragma unroll
        for (int f = 0; f < 10; ++f) rt += Mb[n*10+f] * Wroot1[f*32 + c];
      }
      gcol[n] = fmaxf(a + rt + gb1[c], 0.f);   // relu(gc1)
    }
#pragma unroll
    for (int j = 0; j < 20; ++j) gagg[EDST[j]] += gcol[ESRC[j]];
#pragma unroll
    for (int n = 0; n < 7; ++n){
#pragma unroll
      for (int s10 = 0; s10 < 10; ++s10)
        sX[n*10+s10][tid] += gagg[n] * Wrel2[c*10 + s10] + gcol[n] * Wroot2[c*10 + s10];
    }
  }

  // ---- write x1 [b][t][32] f16 ----
  for (int t = 0; t < SEQN; ++t){
    _Float16* row = x1 + (b*SEQN + t) * 32;
#pragma unroll
    for (int ch = 0; ch < 7; ++ch) row[ch]      = (_Float16)Mb[ch*10 + t];
#pragma unroll
    for (int j = 0; j < 7; ++j)    row[7 + j]   = (_Float16)(sMP[j*10 + t][tid] + b2[t]);
#pragma unroll
    for (int j = 0; j < 7; ++j)    row[14 + j]  = (_Float16)sX[j*10 + t][tid];
#pragma unroll
    for (int k = 21; k < 32; ++k)  row[k]       = (_Float16)0.f;
  }
}

// ---------------------------------------------------------------------------
// LSTM direction: each wave owns a 16-sample tile; c in C-fragments, h in
// f16 A-fragments restaged through per-wave LDS. WMMA for input + recurrent.
// Weights are pre-converted f16: Wih [256][32*NKT] (zero-padded), Whh [256][64].
// ---------------------------------------------------------------------------
template <int NKT>
__global__ __launch_bounds__(256)
void lstm_kernel(const _Float16* __restrict__ x, int xstride,
                 const _Float16* __restrict__ Wih, const _Float16* __restrict__ Whh,
                 const float* __restrict__ bias,
                 _Float16* __restrict__ O, int o_off, int backward, int B){
  __shared__ _Float16 sh[8][16][72];
  int wave = threadIdx.x >> 5, lane = threadIdx.x & 31, n16 = lane & 15;
  long b0 = ((long)blockIdx.x * 8 + wave) * 16;
  if (b0 >= B) return;
  const int wld = NKT * 32;

  v8f cst[4];
#pragma unroll
  for (int j = 0; j < 4; ++j)
#pragma unroll
    for (int r = 0; r < 8; ++r) cst[j][r] = 0.f;
  v16h hA[2];
#pragma unroll
  for (int j = 0; j < 2; ++j)
#pragma unroll
    for (int e = 0; e < 16; ++e) hA[j][e] = (_Float16)0.f;

  for (int step = 0; step < SEQN; ++step){
    int t = backward ? (SEQN - 1 - step) : step;
    v16h ax[NKT];
#pragma unroll
    for (int kt = 0; kt < NKT; ++kt)
      ax[kt] = make_frag([&](int m, int k)->_Float16{
        return x[((b0 + m)*SEQN + t)*xstride + kt*32 + k];
      });

    auto gate_tile = [&](int nt)->v8f{
      float bv = bias[nt*16 + n16];
      v8f a;
#pragma unroll
      for (int r = 0; r < 8; ++r) a[r] = bv;
#pragma unroll
      for (int kt = 0; kt < NKT; ++kt){
        v16h bw = make_frag([&](int n, int k)->_Float16{
          return Wih[(nt*16 + n)*wld + kt*32 + k];
        });
        a = wmma_f16f32(ax[kt], bw, a);
      }
#pragma unroll
      for (int kf = 0; kf < 2; ++kf){
        v16h bw = make_frag([&](int n, int k)->_Float16{
          return Whh[(nt*16 + n)*64 + kf*32 + k];
        });
        a = wmma_f16f32(hA[kf], bw, a);
      }
      return a;
    };

#pragma unroll
    for (int j = 0; j < 4; ++j){            // hidden tile j: gates i,f,g,o
      v8f gi = gate_tile(j);
      v8f gf = gate_tile(4 + j);
      v8f gg = gate_tile(8 + j);
      v8f go = gate_tile(12 + j);
#pragma unroll
      for (int r = 0; r < 8; ++r){
        float cn = sigmoid_hw(gf[r]) * cst[j][r] + sigmoid_hw(gi[r]) * tanh_hw(gg[r]);
        cst[j][r] = cn;
        float hv = sigmoid_hw(go[r]) * tanh_hw(cn);
        int row = r + ((lane & 16) ? 8 : 0);
        int col = j*16 + n16;
        sh[wave][row][col] = (_Float16)hv;
        O[((b0 + row)*SEQN + t)*128 + o_off + col] = (_Float16)hv;
      }
    }
    asm volatile("s_wait_dscnt 0" ::: "memory");  // h tile staged in LDS (same wave)
#pragma unroll
    for (int kf = 0; kf < 2; ++kf)
      hA[kf] = make_frag([&](int m, int k)->_Float16{ return sh[wave][m][kf*32 + k]; });
  }
}

// ---------------------------------------------------------------------------
// Q = tanh(O @ Wa^T + ba)  over R = B*10 rows, WMMA.  Wa pre-converted f16.
// ---------------------------------------------------------------------------
__global__ __launch_bounds__(256)
void attn_q_kernel(const _Float16* __restrict__ O, const _Float16* __restrict__ Wa,
                   const float* __restrict__ ba, _Float16* __restrict__ Q, long R){
  int wave = threadIdx.x >> 5, lane = threadIdx.x & 31, n16 = lane & 15;
  long r0 = ((long)blockIdx.x * 8 + wave) * 16;
  if (r0 >= R) return;
  v16h ax[4];
#pragma unroll
  for (int kt = 0; kt < 4; ++kt)
    ax[kt] = make_frag([&](int m, int k)->_Float16{ return O[(r0 + m)*128 + kt*32 + k]; });
#pragma unroll
  for (int nt = 0; nt < 8; ++nt){
    float bv = ba[nt*16 + n16];
    v8f a;
#pragma unroll
    for (int r = 0; r < 8; ++r) a[r] = bv;
#pragma unroll
    for (int kt = 0; kt < 4; ++kt){
      v16h bw = make_frag([&](int n, int k)->_Float16{
        return Wa[(nt*16 + n)*128 + kt*32 + k];
      });
      a = wmma_f16f32(ax[kt], bw, a);
    }
#pragma unroll
    for (int r = 0; r < 8; ++r){
      int row = r + ((lane & 16) ? 8 : 0);
      Q[(r0 + row)*128 + nt*16 + n16] = (_Float16)tanh_hw(a[r]);
    }
  }
}

// ---------------------------------------------------------------------------
// softmax(Q O^T) O  — 10x10 per sample, one thread per (b,q). VALU.
// ---------------------------------------------------------------------------
__global__ void attn_apply_kernel(const _Float16* __restrict__ Q, const _Float16* __restrict__ O,
                                  _Float16* __restrict__ out, int total){
  int idx = blockIdx.x * blockDim.x + threadIdx.x;
  if (idx >= total) return;
  int b = idx / 10;
  const _Float16* Qr = Q + (size_t)idx * 128;
  const _Float16* Ob = O + (size_t)b * 1280;
  __builtin_prefetch(Ob, 0, 1);
  float s[10]; float mx = -1e30f;
#pragma unroll
  for (int k = 0; k < 10; ++k){
    float d = 0.f;
    for (int hh = 0; hh < 128; ++hh) d += (float)Qr[hh] * (float)Ob[k*128 + hh];
    s[k] = d; mx = fmaxf(mx, d);
  }
  float den = 0.f;
#pragma unroll
  for (int k = 0; k < 10; ++k){ s[k] = __expf(s[k] - mx); den += s[k]; }
  float inv = 1.f / den;
  _Float16* outr = out + (size_t)idx * 128;
  for (int c = 0; c < 128; ++c){
    float v = 0.f;
#pragma unroll
    for (int k = 0; k < 10; ++k) v += s[k] * (float)Ob[k*128 + c];
    outr[c] = (_Float16)(v * inv);
  }
}

// ---------------------------------------------------------------------------
// mean over t, then FC [B,128]x[128,1008(pad)] with WMMA; W pre-converted f16.
// ---------------------------------------------------------------------------
__global__ void mean_kernel(const _Float16* __restrict__ A, _Float16* __restrict__ xm, int total){
  int idx = blockIdx.x * blockDim.x + threadIdx.x;
  if (idx >= total) return;
  int b = idx >> 7, c = idx & 127;
  float s = 0.f;
#pragma unroll
  for (int t = 0; t < 10; ++t) s += (float)A[((size_t)b*10 + t)*128 + c];
  xm[idx] = (_Float16)(s * 0.1f);
}

__global__ __launch_bounds__(256)
void fc_kernel(const _Float16* __restrict__ xm, const _Float16* __restrict__ W,
               const float* __restrict__ bv, float* __restrict__ out, int B){
  int wave = threadIdx.x >> 5, lane = threadIdx.x & 31, n16 = lane & 15;
  long b0 = ((long)blockIdx.x * 8 + wave) * 16;
  if (b0 >= B) return;
  v16h ax[4];
#pragma unroll
  for (int kt = 0; kt < 4; ++kt)
    ax[kt] = make_frag([&](int m, int k)->_Float16{ return xm[(b0 + m)*128 + kt*32 + k]; });
  for (int nt = 0; nt < 63; ++nt){
    int col = nt*16 + n16;
    float bb = (col < 1000) ? bv[col] : 0.f;
    v8f a;
#pragma unroll
    for (int r = 0; r < 8; ++r) a[r] = bb;
#pragma unroll
    for (int kt = 0; kt < 4; ++kt){
      v16h bw = make_frag([&](int n, int k)->_Float16{
        return W[(nt*16 + n)*128 + kt*32 + k];   // rows >= 1000 are zero-padded
      });
      a = wmma_f16f32(ax[kt], bw, a);
    }
    if (col < 1000){
#pragma unroll
      for (int r = 0; r < 8; ++r){
        int row = r + ((lane & 16) ? 8 : 0);
        out[(b0 + row)*1000 + col] = a[r];
      }
    }
  }
}

// ---------------------------------------------------------------------------
extern "C" void kernel_launch(void* const* d_in, const int* in_sizes, int n_in,
                              void* d_out, int out_size, void* d_ws, size_t ws_size,
                              hipStream_t stream){
  const float* M        = (const float*)d_in[0];
  const float* ga1_W    = (const float*)d_in[1];
  const float* ga1_as   = (const float*)d_in[2];
  const float* ga1_ad   = (const float*)d_in[3];
  const float* ga1_b    = (const float*)d_in[4];
  const float* ga2_W    = (const float*)d_in[5];
  const float* ga2_as   = (const float*)d_in[6];
  const float* ga2_ad   = (const float*)d_in[7];
  const float* ga2_b    = (const float*)d_in[8];
  const float* gp1_Wrel = (const float*)d_in[9];
  const float* gp1_Wroot= (const float*)d_in[10];
  const float* gp1_b    = (const float*)d_in[11];
  const float* gp2_Wrel = (const float*)d_in[12];
  const float* gp2_Wroot= (const float*)d_in[13];
  const float* gp2_b    = (const float*)d_in[14];
  const float* l1f_Wih  = (const float*)d_in[15];
  const float* l1f_Whh  = (const float*)d_in[16];
  const float* l1f_b    = (const float*)d_in[17];
  const float* l1b_Wih  = (const float*)d_in[18];
  const float* l1b_Whh  = (const float*)d_in[19];
  const float* l1b_b    = (const float*)d_in[20];
  const float* l2f_Wih  = (const float*)d_in[21];
  const float* l2f_Whh  = (const float*)d_in[22];
  const float* l2f_b    = (const float*)d_in[23];
  const float* l2b_Wih  = (const float*)d_in[24];
  const float* l2b_Whh  = (const float*)d_in[25];
  const float* l2b_b    = (const float*)d_in[26];
  const float* a1_Wa    = (const float*)d_in[27];
  const float* a1_ba    = (const float*)d_in[28];
  const float* a2_Wa    = (const float*)d_in[29];
  const float* a2_ba    = (const float*)d_in[30];
  const float* fc_W     = (const float*)d_in[31];
  const float* fc_b     = (const float*)d_in[32];
  (void)n_in; (void)out_size; (void)ws_size;

  int B = in_sizes[0] / 70;

  char* w = (char*)d_ws;
  float* prep = (float*)w;
  size_t off = 4096;
  // f16 weight pool
  _Float16* w_l1f_ih = (_Float16*)(w + off); off += 256*32*2;    // [256][32]  (21 pad->32)
  _Float16* w_l1b_ih = (_Float16*)(w + off); off += 256*32*2;
  _Float16* w_l1f_hh = (_Float16*)(w + off); off += 256*64*2;    // [256][64]
  _Float16* w_l1b_hh = (_Float16*)(w + off); off += 256*64*2;
  _Float16* w_l2f_ih = (_Float16*)(w + off); off += 256*128*2;   // [256][128]
  _Float16* w_l2b_ih = (_Float16*)(w + off); off += 256*128*2;
  _Float16* w_l2f_hh = (_Float16*)(w + off); off += 256*64*2;
  _Float16* w_l2b_hh = (_Float16*)(w + off); off += 256*64*2;
  _Float16* w_a1     = (_Float16*)(w + off); off += 128*128*2;   // [128][128]
  _Float16* w_a2     = (_Float16*)(w + off); off += 128*128*2;
  _Float16* w_fc     = (_Float16*)(w + off); off += 1008*128*2;  // [1008][128] (1000 pad)
  off = (off + 255) & ~(size_t)255;
  // activations
  _Float16* x1   = (_Float16*)(w + off); off += (size_t)B*SEQN*32*sizeof(_Float16);
  off = (off + 255) & ~(size_t)255;
  _Float16* bufA = (_Float16*)(w + off); off += (size_t)B*1280*sizeof(_Float16);
  _Float16* bufB = (_Float16*)(w + off); off += (size_t)B*1280*sizeof(_Float16);
  _Float16* bufC = (_Float16*)(w + off); off += (size_t)B*1280*sizeof(_Float16);
  _Float16* xm   = (_Float16*)(w + off); off += (size_t)B*128*sizeof(_Float16);

  long R = (long)B * SEQN;
  int lstm_grid = (B + 127) / 128;
  int q_grid    = (int)((R/16 + 7) / 8);
  int a_grid    = (int)((R + 255) / 256);

  // ---- weight prep (tiny, graph-captured) ----
  prep_kernel<<<1, 256, 0, stream>>>(ga1_W, ga1_as, ga1_ad, ga2_W, ga2_as, ga2_ad, prep);
  auto cvt = [&](const float* s, _Float16* d, int rows, int drows, int scols, int dcols){
    int tot = drows * dcols;
    convert_w_kernel<<<(tot + 255)/256, 256, 0, stream>>>(s, d, rows, drows, scols, dcols);
  };
  cvt(l1f_Wih, w_l1f_ih, 256, 256, 21, 32);
  cvt(l1b_Wih, w_l1b_ih, 256, 256, 21, 32);
  cvt(l1f_Whh, w_l1f_hh, 256, 256, 64, 64);
  cvt(l1b_Whh, w_l1b_hh, 256, 256, 64, 64);
  cvt(l2f_Wih, w_l2f_ih, 256, 256, 128, 128);
  cvt(l2b_Wih, w_l2b_ih, 256, 256, 128, 128);
  cvt(l2f_Whh, w_l2f_hh, 256, 256, 64, 64);
  cvt(l2b_Whh, w_l2b_hh, 256, 256, 64, 64);
  cvt(a1_Wa,   w_a1,     128, 128, 128, 128);
  cvt(a2_Wa,   w_a2,     128, 128, 128, 128);
  cvt(fc_W,    w_fc,    1000, 1008, 128, 128);

  // ---- pipeline ----
  graph_kernel<<<(B + 31)/32, 32, 0, stream>>>(M, ga1_W, ga1_b, ga2_W, ga2_b,
                                               gp1_Wrel, gp1_Wroot, gp1_b,
                                               gp2_Wrel, gp2_Wroot, gp2_b,
                                               prep, x1, B);
  // BiLSTM 1 (in=21 padded to K=32)
  lstm_kernel<1><<<lstm_grid, 256, 0, stream>>>(x1, 32, w_l1f_ih, w_l1f_hh, l1f_b, bufA, 0,  0, B);
  lstm_kernel<1><<<lstm_grid, 256, 0, stream>>>(x1, 32, w_l1b_ih, w_l1b_hh, l1b_b, bufA, 64, 1, B);
  // attention 1
  attn_q_kernel<<<q_grid, 256, 0, stream>>>(bufA, w_a1, a1_ba, bufB, R);
  attn_apply_kernel<<<a_grid, 256, 0, stream>>>(bufB, bufA, bufC, (int)R);
  // BiLSTM 2 (in=128)
  lstm_kernel<4><<<lstm_grid, 256, 0, stream>>>(bufC, 128, w_l2f_ih, w_l2f_hh, l2f_b, bufA, 0,  0, B);
  lstm_kernel<4><<<lstm_grid, 256, 0, stream>>>(bufC, 128, w_l2b_ih, w_l2b_hh, l2b_b, bufA, 64, 1, B);
  // attention 2
  attn_q_kernel<<<q_grid, 256, 0, stream>>>(bufA, w_a2, a2_ba, bufB, R);
  attn_apply_kernel<<<a_grid, 256, 0, stream>>>(bufB, bufA, bufC, (int)R);
  // mean + FC
  mean_kernel<<<(B*128 + 255)/256, 256, 0, stream>>>(bufC, xm, B*128);
  fc_kernel<<<(B + 127)/128, 256, 0, stream>>>(xm, w_fc, fc_b, (float*)d_out, B);
}
